// Attention_40372692582854
// MI455X (gfx1250) — matrix operand order repaired and checked by reading, not compiled, not run on hardware
//
#include <hip/hip_runtime.h>
#include <hip/hip_bf16.h>
#include <math.h>

typedef __attribute__((ext_vector_type(16))) _Float16 v16h;
typedef __attribute__((ext_vector_type(8)))  _Float16 v8h;
typedef __attribute__((ext_vector_type(2)))  _Float16 v2h;
typedef __attribute__((ext_vector_type(8)))  float    v8f;
typedef int v4i_ls __attribute__((vector_size(16)));

#define D_IN  768
#define S_LEN 2048
#define B_N   8
#define M_TOT (B_N * S_LEN)   /* 16384 */
#define N3D   (3 * D_IN)      /* 2304  */

// ---- CDNA5 async global->LDS path (guarded so both toolchains compile) ----
#if __has_builtin(__builtin_amdgcn_global_load_async_to_lds_b128)
#define HAVE_ASYNC_LDS 1
#define GLOBAL_AS_V4(p) ((__attribute__((address_space(1))) v4i_ls*)(size_t)(p))
#define LDS_AS_V4(p)    ((__attribute__((address_space(3))) v4i_ls*)(unsigned)(size_t)(p))
#if __has_builtin(__builtin_amdgcn_s_wait_asynccnt)
#define ASYNC_WAIT() __builtin_amdgcn_s_wait_asynccnt(0)
#else
#define ASYNC_WAIT() asm volatile("s_wait_asynccnt 0x0" ::: "memory")
#endif
#else
#define HAVE_ASYNC_LDS 0
#endif

static __device__ __forceinline__ v16h cat16(v8h lo, v8h hi) {
  return __builtin_shufflevector(lo, hi, 0,1,2,3,4,5,6,7,8,9,10,11,12,13,14,15);
}
static __device__ __forceinline__ v8f wmma_f16f32(v16h a, v16h b, v8f c) {
  // (neg_a, A, neg_b, B, c_mod, C, reuse_a, reuse_b)
  return __builtin_amdgcn_wmma_f32_16x16x32_f16(false, a, false, b, (short)0, c, false, false);
}

// ---------------------------------------------------------------------------
// f32 -> f16 elementwise convert (weights). n must be a multiple of 256.
// ---------------------------------------------------------------------------
__global__ __launch_bounds__(256) void cvt_f32_f16(const float* __restrict__ src,
                                                   _Float16* __restrict__ dst, int n) {
  int i = blockIdx.x * 256 + threadIdx.x;
  if (i < n) dst[i] = (_Float16)src[i];
}

// ---------------------------------------------------------------------------
// LayerNorm over D=768, fp32 stats, f16 output. One row per 256-thread block.
// ---------------------------------------------------------------------------
__global__ __launch_bounds__(256) void layernorm_kernel(const float* __restrict__ x,
                                                        const float* __restrict__ gamma,
                                                        const float* __restrict__ beta,
                                                        _Float16* __restrict__ x1h) {
  __shared__ float sh1[256], sh2[256];
  int row = blockIdx.x;
  int t = threadIdx.x;
  const float* xr = x + (size_t)row * D_IN;
  float v0 = xr[t], v1 = xr[t + 256], v2 = xr[t + 512];
  sh1[t] = v0 + v1 + v2;
  sh2[t] = v0 * v0 + v1 * v1 + v2 * v2;
  __syncthreads();
  for (int off = 128; off > 0; off >>= 1) {
    if (t < off) { sh1[t] += sh1[t + off]; sh2[t] += sh2[t + off]; }
    __syncthreads();
  }
  float mu  = sh1[0] * (1.0f / D_IN);
  float var = sh2[0] * (1.0f / D_IN) - mu * mu;
  float rstd = rsqrtf(var + 1e-5f);
  _Float16* o = x1h + (size_t)row * D_IN;
  o[t]       = (_Float16)((v0 - mu) * rstd * gamma[t]       + beta[t]);
  o[t + 256] = (_Float16)((v1 - mu) * rstd * gamma[t + 256] + beta[t + 256]);
  o[t + 512] = (_Float16)((v2 - mu) * rstd * gamma[t + 512] + beta[t + 512]);
}

// ---------------------------------------------------------------------------
// Generic f16 x f16 -> f32 WMMA GEMM. Block tile 64(M) x 128(N), K-step 32.
// 256 threads = 8 waves; wave (wm,wn) owns a 16x64 slab => 4 accumulators.
// A tile is DMA'd to LDS with GLOBAL_LOAD_ASYNC_TO_LDS_B128 (ASYNCcnt) when
// available; B tile is staged TRANSPOSED in LDS (Bs[col][k]) via packed
// ds_store_b32 of (K,K+1) f16 pairs so per-column B fragments are contiguous
// ds_load_b128s (matches wave32 WMMA B layout). global_prefetch_b8 warms the
// next K-step.
// mode 0: C_h = A*B + bias                  (f16 out)
// mode 1: C_h = A*B                         (f16 out)
// mode 2: C_f = gelu(A*B + bias + resid)    (f32 out, exact erf GELU)
// Batched via blockIdx.z with element strides sA/sB/sC.
// ---------------------------------------------------------------------------
__global__ __launch_bounds__(256) void gemm_f16(const _Float16* __restrict__ A, long long sA, int lda,
                                                const _Float16* __restrict__ B, long long sB, int ldb,
                                                _Float16* __restrict__ Ch, float* __restrict__ Cf,
                                                long long sC, int ldc,
                                                const float* __restrict__ bias,
                                                const float* __restrict__ resid,
                                                int K, int mode) {
  __shared__ _Float16 As[64 * 32];
  __shared__ _Float16 Bs[128 * 32];

  int z = blockIdx.z;
  A += (size_t)z * sA;
  B += (size_t)z * sB;
  if (Ch) Ch += (size_t)z * sC;
  if (Cf) Cf += (size_t)z * sC;

  int bm = blockIdx.y * 64;
  int bn = blockIdx.x * 128;
  int t = threadIdx.x, lane = t & 31, w = t >> 5;
  int wm = w & 3, wn = w >> 2;
  int mrow = lane & 15;
  int hi = lane >> 4;

  v8f zero = {0.f, 0.f, 0.f, 0.f, 0.f, 0.f, 0.f, 0.f};
  v8f acc[4];
  acc[0] = zero; acc[1] = zero; acc[2] = zero; acc[3] = zero;

  int arow = t >> 2, aseg = t & 3;   // A staging: 64 rows x 4 segs of 8 halves
  int rp   = t >> 4, cs   = t & 15;  // B staging: 16 row-pairs x 16 col segs of 8

  for (int k0 = 0; k0 < K; k0 += 32) {
    const _Float16* asrc  = A + (size_t)(bm + arow) * lda + k0 + aseg * 8;
    const _Float16* bsrc0 = B + (size_t)(k0 + rp * 2) * ldb + bn + cs * 8;
    const _Float16* bsrc1 = bsrc0 + ldb;

    // Warm L2/L0 for next K-step (speculative; OOB prefetch silently dropped).
    __builtin_prefetch(asrc + 32, 0, 3);
    __builtin_prefetch(bsrc0 + (size_t)32 * ldb, 0, 3);

    // ---- stage A tile (row-major), async DMA straight into LDS ----
#if HAVE_ASYNC_LDS
    __builtin_amdgcn_global_load_async_to_lds_b128(
        GLOBAL_AS_V4(asrc), LDS_AS_V4(&As[arow * 32 + aseg * 8]), 0, 0);
#else
    *(v8h*)&As[arow * 32 + aseg * 8] = *(const v8h*)asrc;
#endif

    // ---- stage B tile transposed: Bs[col][k], packed (k,k+1) b32 stores ----
    v8h b0 = *(const v8h*)bsrc0;
    v8h b1 = *(const v8h*)bsrc1;
#pragma unroll
    for (int i = 0; i < 8; ++i) {
      v2h pr;
      pr.x = b0[i];
      pr.y = b1[i];
      *(v2h*)&Bs[(cs * 8 + i) * 32 + rp * 2] = pr;
    }

#if HAVE_ASYNC_LDS
    ASYNC_WAIT();
#endif
    __syncthreads();

    // ---- A fragment: lane-half K interleave 0-7/16-23 vs 8-15/24-31 ----
    int kA0 = hi * 8;
    v8h alo = *(const v8h*)&As[(wm * 16 + mrow) * 32 + kA0];
    v8h ahi = *(const v8h*)&As[(wm * 16 + mrow) * 32 + kA0 + 16];
    v16h af = cat16(alo, ahi);

#pragma unroll
    for (int j = 0; j < 4; ++j) {
      int col = wn * 64 + j * 16 + mrow;       // N per lane
      int kB0 = hi * 8;                        // K interleave per lane-half
      v8h bl = *(const v8h*)&Bs[col * 32 + kB0];
      v8h bh = *(const v8h*)&Bs[col * 32 + kB0 + 16];
      acc[j] = wmma_f16f32(af, cat16(bl, bh), acc[j]);
    }
    __syncthreads();
  }

  // ---- epilogue ----
#pragma unroll
  for (int j = 0; j < 4; ++j) {
    int gcol = bn + wn * 64 + j * 16 + mrow;
#pragma unroll
    for (int r = 0; r < 8; ++r) {
      int grow = bm + wm * 16 + r + hi * 8;
      float v = acc[j][r];
      if (mode == 0) {
        Ch[(size_t)grow * ldc + gcol] = (_Float16)(v + bias[gcol]);
      } else if (mode == 1) {
        Ch[(size_t)grow * ldc + gcol] = (_Float16)v;
      } else {
        float y = v + bias[gcol] + resid[(size_t)grow * ldc + gcol];
        Cf[(size_t)grow * ldc + gcol] = 0.5f * y * (1.0f + erff(y * 0.70710678118654752f));
      }
    }
  }
}

// ---------------------------------------------------------------------------
// Attention probabilities (flash-style, two passes over keys).
// Each wave owns one 16-query tile of one batch. Scores via 24 chained
// v_wmma_f32_16x16x32_f16 per 16-key tile, fragments loaded from L2-resident
// global qkv (f16). Pass 0: online row max / sum-exp (half-wave shfl_xor
// reductions). Pass 1: write P = exp(s - m) / l as f16.
// grid = (S/128, B), block = 256 (8 waves).
// ---------------------------------------------------------------------------
__global__ __launch_bounds__(256) void attn_prob_kernel(const _Float16* __restrict__ qkvh,
                                                        _Float16* __restrict__ P,
                                                        float scale) {
  int b = blockIdx.y;
  int w = threadIdx.x >> 5;
  int lane = threadIdx.x & 31;
  int q0 = (blockIdx.x * 8 + w) * 16;
  int mrow = lane & 15;
  int hi = lane >> 4;
  int kA0 = hi * 8;

  const _Float16* qb = qkvh + (size_t)b * S_LEN * N3D;
  const _Float16* qrow = qb + (size_t)(q0 + mrow) * N3D;   // Q at cols [0,768)
  _Float16* Pb = P + (size_t)b * S_LEN * S_LEN;

  float m[8], l[8];
#pragma unroll
  for (int r = 0; r < 8; ++r) { m[r] = -1e30f; l[r] = 0.0f; }

  for (int phase = 0; phase < 2; ++phase) {
    if (phase == 1) {
#pragma unroll
      for (int r = 0; r < 8; ++r) l[r] = 1.0f / l[r];
    }
    for (int k0 = 0; k0 < S_LEN; k0 += 16) {
      // B fragment: N = key (lane), K = channel contiguous. K at cols [768,1536)
      const _Float16* krow = qb + (size_t)(k0 + mrow) * N3D + D_IN + hi * 8;
      v8f s = {};
#pragma unroll
      for (int kt = 0; kt < D_IN / 32; ++kt) {
        v8h alo = *(const v8h*)(qrow + kt * 32 + kA0);
        v8h ahi = *(const v8h*)(qrow + kt * 32 + kA0 + 16);
        v8h blo = *(const v8h*)(krow + kt * 32);
        v8h bhi = *(const v8h*)(krow + kt * 32 + 16);
        s = wmma_f16f32(cat16(alo, ahi), cat16(blo, bhi), s);
      }
      if (phase == 0) {
#pragma unroll
        for (int r = 0; r < 8; ++r) {
          float v = s[r] * scale;
          float tmax = v;
          tmax = fmaxf(tmax, __shfl_xor(tmax, 1, 32));
          tmax = fmaxf(tmax, __shfl_xor(tmax, 2, 32));
          tmax = fmaxf(tmax, __shfl_xor(tmax, 4, 32));
          tmax = fmaxf(tmax, __shfl_xor(tmax, 8, 32));
          float mn = fmaxf(m[r], tmax);
          float p = __expf(v - mn);
          float ts = p;
          ts += __shfl_xor(ts, 1, 32);
          ts += __shfl_xor(ts, 2, 32);
          ts += __shfl_xor(ts, 4, 32);
          ts += __shfl_xor(ts, 8, 32);
          l[r] = l[r] * __expf(m[r] - mn) + ts;
          m[r] = mn;
        }
      } else {
#pragma unroll
        for (int r = 0; r < 8; ++r) {
          float p = __expf(s[r] * scale - m[r]) * l[r];   // l holds 1/sum
          int row = q0 + r + hi * 8;
          Pb[(size_t)row * S_LEN + k0 + mrow] = (_Float16)p;
        }
      }
    }
  }
}

// ---------------------------------------------------------------------------
// Host-side orchestration. Workspace layout (256B-aligned, ~189 MB total):
//   x1h   [16384 x  768] f16   LN output
//   wqkvh [  768 x 2304] f16
//   wouth [  768 x  768] f16
//   qkvh  [16384 x 2304] f16   QKV projection
//   P     [8 x 2048 x 2048] f16  attention probabilities
//   attnh [16384 x  768] f16   attention output
// ---------------------------------------------------------------------------
extern "C" void kernel_launch(void* const* d_in, const int* in_sizes, int n_in,
                              void* d_out, int out_size, void* d_ws, size_t ws_size,
                              hipStream_t stream) {
  const float* x      = (const float*)d_in[0];
  const float* gamma  = (const float*)d_in[1];
  const float* beta   = (const float*)d_in[2];
  const float* w_qkv  = (const float*)d_in[3];
  const float* b_qkv  = (const float*)d_in[4];
  const float* w_out  = (const float*)d_in[5];
  const float* b_out  = (const float*)d_in[6];
  float* out = (float*)d_out;

  const size_t off_x1h   = 0;
  const size_t off_wqkvh = off_x1h   + (size_t)M_TOT * D_IN * 2;
  const size_t off_wouth = off_wqkvh + (size_t)D_IN * N3D * 2;
  const size_t off_qkvh  = off_wouth + (size_t)D_IN * D_IN * 2;
  const size_t off_P     = off_qkvh  + (size_t)M_TOT * N3D * 2;
  const size_t off_attnh = off_P     + (size_t)B_N * S_LEN * S_LEN * 2;

  char* ws = (char*)d_ws;
  _Float16* x1h   = (_Float16*)(ws + off_x1h);
  _Float16* wqkvh = (_Float16*)(ws + off_wqkvh);
  _Float16* wouth = (_Float16*)(ws + off_wouth);
  _Float16* qkvh  = (_Float16*)(ws + off_qkvh);
  _Float16* P     = (_Float16*)(ws + off_P);
  _Float16* attnh = (_Float16*)(ws + off_attnh);

  // 1) weights -> f16
  cvt_f32_f16<<<dim3((D_IN * N3D) / 256), 256, 0, stream>>>(w_qkv, wqkvh, D_IN * N3D);
  cvt_f32_f16<<<dim3((D_IN * D_IN) / 256), 256, 0, stream>>>(w_out, wouth, D_IN * D_IN);

  // 2) LayerNorm
  layernorm_kernel<<<dim3(M_TOT), 256, 0, stream>>>(x, gamma, beta, x1h);

  // 3) QKV GEMM: [16384,768] x [768,2304] + bias -> f16
  gemm_f16<<<dim3(N3D / 128, M_TOT / 64, 1), 256, 0, stream>>>(
      x1h, 0, D_IN, wqkvh, 0, N3D, qkvh, nullptr, 0, N3D, b_qkv, nullptr, D_IN, 0);

  // 4) attention probabilities (softmax(QK^T * scale)) -> f16
  attn_prob_kernel<<<dim3(S_LEN / 128, B_N), 256, 0, stream>>>(
      qkvh, P, 1.0f / sqrtf((float)D_IN));

  // 5) PV GEMM per batch: [2048,2048] x [2048,768] -> f16 (V = qkv cols [1536,2304))
  gemm_f16<<<dim3(D_IN / 128, S_LEN / 64, B_N), 256, 0, stream>>>(
      P, (long long)S_LEN * S_LEN, S_LEN,
      qkvh + 2 * D_IN, (long long)S_LEN * N3D, N3D,
      attnh, nullptr, (long long)S_LEN * D_IN, D_IN, nullptr, nullptr, S_LEN, 1);

  // 6) output projection + bias + residual + exact GELU -> f32 d_out
  gemm_f16<<<dim3(D_IN / 128, M_TOT / 64, 1), 256, 0, stream>>>(
      attnh, 0, D_IN, wouth, 0, D_IN, nullptr, out, 0, D_IN, b_out, x, D_IN, 2);
}
